// GaGCN_identity_18013092839705
// MI455X (gfx1250) — compile-verified, matrix operand order checked by reading, not run on hardware
//
#include <hip/hip_runtime.h>
#include <hip/hip_bf16.h>
#include <math.h>

// ---------------- types ----------------
typedef __bf16 v16bf __attribute__((ext_vector_type(16)));
typedef __bf16 v8bf  __attribute__((ext_vector_type(8)));
typedef float  v8f   __attribute__((ext_vector_type(8)));
typedef float  v4f   __attribute__((ext_vector_type(4)));

#define DD   256
#define C1   128
#define C2   256
#define BB   8
#define NK1  128
#define NK2  64
#define WPAD 264   // padded LDS row stride in bf16 (528 B -> conflict-free)

__device__ __forceinline__ float lrelu(float v) { return v > 0.f ? v : 0.33f * v; }
__device__ __forceinline__ float sigm(float v)  { return 1.f / (1.f + expf(-v)); }

// ---------------- stage 0: layer-1 E2E (a1[b,c,i], b1[b,c,j]) ----------------
__global__ void __launch_bounds__(256) k_e2e1(
    const float* __restrict__ x,
    const float* __restrict__ w1a, const float* __restrict__ b1a,
    const float* __restrict__ w1b, const float* __restrict__ b1b,
    float* __restrict__ a1, float* __restrict__ b1)
{
  int gid = blockIdx.x * blockDim.x + threadIdx.x;
  const int half = BB * C1 * DD;
  bool isB = gid >= half;
  int t = isB ? gid - half : gid;
  int i = t % DD;
  int c = (t / DD) % C1;
  int b = t / (DD * C1);
  float acc = 0.f;
  if (!isB) {
    const float* xr = x + (size_t)b * DD * DD + (size_t)i * DD;
    const float* wr = w1a + (size_t)c * DD;
    for (int j = 0; j < DD; j += 4) {
      v4f xv = *(const v4f*)(xr + j);
      v4f wv = *(const v4f*)(wr + j);
      acc += xv[0]*wv[0] + xv[1]*wv[1] + xv[2]*wv[2] + xv[3]*wv[3];
    }
    a1[t] = acc + b1a[c];
  } else {
    const float* xc = x + (size_t)b * DD * DD + i;     // column i, stride DD
    const float* wr = w1b + (size_t)c * DD;
    for (int r = 0; r < DD; ++r) acc += xc[(size_t)r * DD] * wr[r];
    b1[t] = acc + b1b[c];
  }
}

// ---------------- bf16 weight conversion ----------------
__global__ void __launch_bounds__(256) k_cvt_bf16(
    const float* __restrict__ s, __bf16* __restrict__ d, int n)
{
  int i = blockIdx.x * blockDim.x + threadIdx.x;
  if (i < n) d[i] = (__bf16)s[i];
}

// e2n weights padded 8 -> 16 output channels (zeros)
__global__ void __launch_bounds__(256) k_cvt_e2n(
    const float* __restrict__ s, __bf16* __restrict__ d)
{
  int i = blockIdx.x * blockDim.x + threadIdx.x;   // 16 * 65536
  int p = i >> 16;
  d[i] = (p < 8) ? (__bf16)s[i] : (__bf16)0.f;
}

// ---------------- stage 1: layer-2 E2E via WMMA, LDS-staged weights ----------------
// out[b,p,r] = sum_{c,k} lrelu(rowv[b,c,r] + colv[b,c,k]) * W[p, c*DD+k] + bias[p]
// Block: 128 rows x 64 cols, 8 waves (each 16 rows x 64 cols, 4 accumulators).
// Per channel c: stage W[n0..n0+63][c*DD .. +255] (32 KB bf16, padded) and the
// colv row (1 KB) into double-buffered LDS; all fragment reads come from LDS.
__global__ void __launch_bounds__(256) k_e2e2_wmma(
    const float* __restrict__ a1, const float* __restrict__ b1,
    const __bf16* __restrict__ w2a_bf, const __bf16* __restrict__ w2b_bf,
    const float* __restrict__ b2a, const float* __restrict__ b2b,
    float* __restrict__ a2, float* __restrict__ b2)
{
  const int C = C1;
  const size_t K = (size_t)C * DD;                  // 32768
  extern __shared__ char smem[];
  __bf16 (*Wt)[64][WPAD] = (__bf16(*)[64][WPAD])smem;                 // 2 bufs
  float* colb = (float*)(smem + (size_t)2 * 64 * WPAD * sizeof(__bf16));

  int tile  = blockIdx.x;                           // 0..7
  int b     = blockIdx.y;
  int which = blockIdx.z;

  const float*  rowv = which ? b1 : a1;
  const float*  colv = which ? a1 : b1;
  const __bf16* W    = which ? w2b_bf : w2a_bf;
  const float*  bias = which ? b2b : b2a;
  float*        out  = which ? b2 : a2;

  int ti = tile & 1, tn = tile >> 1;
  int i0 = ti * 128, n0 = tn * 64;
  int wave = threadIdx.x >> 5, lane = threadIdx.x & 31;
  int i_base = i0 + wave * 16;
  int m  = lane & 15;
  int kb = (lane >> 4) << 3;                        // 0 or 8

  int sr = threadIdx.x & 63;                        // staging row
  int ss = threadIdx.x >> 6;                        // staging segment (64 bf16 each)
  const float* colp = colv + (size_t)b * C * DD;
  const float* rowp = rowv + (size_t)b * C * DD + i_base + m;
  const __bf16* wsrc0 = W + (size_t)(n0 + sr) * K + (size_t)ss * 64;

  // prologue: stage channel 0
  {
    const v8bf* src = (const v8bf*)wsrc0;
    v8bf* dst = (v8bf*)(&Wt[0][sr][ss * 64]);
#pragma unroll
    for (int u = 0; u < 8; ++u) dst[u] = src[u];
    colb[threadIdx.x] = colp[threadIdx.x];
  }
  __syncthreads();

  v8f acc[4] = {{}, {}, {}, {}};
  for (int c = 0; c < C; ++c) {
    int cur = c & 1;
    if (c + 1 < C) {                                // prefetch next channel
      const v8bf* src = (const v8bf*)(wsrc0 + (size_t)(c + 1) * DD);
      v8bf* dst = (v8bf*)(&Wt[cur ^ 1][sr][ss * 64]);
#pragma unroll
      for (int u = 0; u < 8; ++u) dst[u] = src[u];
      colb[(cur ^ 1) * 256 + threadIdx.x] = colp[(size_t)(c + 1) * DD + threadIdx.x];
    }
    float av = rowp[(size_t)c * DD];
    const float* cb = colb + cur * 256 + kb;
    for (int j0 = 0; j0 < DD; j0 += 32) {
      v4f c0 = *(const v4f*)(cb + j0);
      v4f c1 = *(const v4f*)(cb + j0 + 4);
      v4f c2 = *(const v4f*)(cb + j0 + 16);
      v4f c3 = *(const v4f*)(cb + j0 + 20);
      v16bf af;
#pragma unroll
      for (int e = 0; e < 4; ++e) af[e]      = (__bf16)lrelu(av + c0[e]);
#pragma unroll
      for (int e = 0; e < 4; ++e) af[4 + e]  = (__bf16)lrelu(av + c1[e]);
#pragma unroll
      for (int e = 0; e < 4; ++e) af[8 + e]  = (__bf16)lrelu(av + c2[e]);
#pragma unroll
      for (int e = 0; e < 4; ++e) af[12 + e] = (__bf16)lrelu(av + c3[e]);
#pragma unroll
      for (int q = 0; q < 4; ++q) {
        union { v16bf v; v8bf h[2]; } bf;
        bf.h[0] = *(const v8bf*)(&Wt[cur][m + 16 * q][j0 + kb]);
        bf.h[1] = *(const v8bf*)(&Wt[cur][m + 16 * q][j0 + kb + 16]);
        acc[q] = __builtin_amdgcn_wmma_f32_16x16x32_bf16(false, af, false, bf.v,
                                                         (short)0, acc[q], false, false);
      }
    }
    __syncthreads();
  }

  // C/D layout: lane = N (+16 for M>=8), VGPR r -> M = r + (lane>=16 ? 8 : 0)
  int rbase = (lane >> 4) * 8;
#pragma unroll
  for (int q = 0; q < 4; ++q) {
    int p = n0 + m + 16 * q;
    float bv = bias[p];
    float* o = out + ((size_t)b * DD + p) * DD + i_base + rbase;
    *(v4f*)o       = (v4f){acc[q][0] + bv, acc[q][1] + bv, acc[q][2] + bv, acc[q][3] + bv};
    *(v4f*)(o + 4) = (v4f){acc[q][4] + bv, acc[q][5] + bv, acc[q][6] + bv, acc[q][7] + bv};
  }
}

// ---------------- stage 2: E2N via WMMA (N padded to 16), LDS-staged ----------------
// nodes[b,i,p] = lrelu( sum_{c,j} lrelu(a2[b,c,i]+b2[b,c,j]) * e2n[p,c,j] + e2n_b[p] )
__global__ void __launch_bounds__(256) k_e2n_wmma(
    const float* __restrict__ a2, const float* __restrict__ b2,
    const __bf16* __restrict__ e2n_bf, const float* __restrict__ e2n_b,
    float* __restrict__ nodes)
{
  const int C = C2;
  const size_t K = (size_t)C * DD;                  // 65536
  __shared__ __bf16 Wt[2][16][WPAD];
  __shared__ float  colb[2][256];

  int b = blockIdx.y;
  int wave = threadIdx.x >> 5, lane = threadIdx.x & 31;
  int i_base = blockIdx.x * 128 + wave * 16;
  int m  = lane & 15;
  int kb = (lane >> 4) << 3;

  int sr = threadIdx.x & 15;                        // staging row (16 rows)
  int ss = threadIdx.x >> 4;                        // 0..15, 16 bf16 each
  const float* colp = b2 + (size_t)b * C * DD;
  const float* rowp = a2 + (size_t)b * C * DD + i_base + m;
  const __bf16* wsrc0 = e2n_bf + (size_t)sr * K + (size_t)ss * 16;

  {
    const v8bf* src = (const v8bf*)wsrc0;
    v8bf* dst = (v8bf*)(&Wt[0][sr][ss * 16]);
    dst[0] = src[0]; dst[1] = src[1];
    colb[0][threadIdx.x] = colp[threadIdx.x];
  }
  __syncthreads();

  v8f acc = {};
  for (int c = 0; c < C; ++c) {
    int cur = c & 1;
    if (c + 1 < C) {
      const v8bf* src = (const v8bf*)(wsrc0 + (size_t)(c + 1) * DD);
      v8bf* dst = (v8bf*)(&Wt[cur ^ 1][sr][ss * 16]);
      dst[0] = src[0]; dst[1] = src[1];
      colb[cur ^ 1][threadIdx.x] = colp[(size_t)(c + 1) * DD + threadIdx.x];
    }
    float av = rowp[(size_t)c * DD];
    const float* cb = &colb[cur][kb];
    for (int j0 = 0; j0 < DD; j0 += 32) {
      v4f c0 = *(const v4f*)(cb + j0);
      v4f c1 = *(const v4f*)(cb + j0 + 4);
      v4f c2 = *(const v4f*)(cb + j0 + 16);
      v4f c3 = *(const v4f*)(cb + j0 + 20);
      v16bf af;
#pragma unroll
      for (int e = 0; e < 4; ++e) af[e]      = (__bf16)lrelu(av + c0[e]);
#pragma unroll
      for (int e = 0; e < 4; ++e) af[4 + e]  = (__bf16)lrelu(av + c1[e]);
#pragma unroll
      for (int e = 0; e < 4; ++e) af[8 + e]  = (__bf16)lrelu(av + c2[e]);
#pragma unroll
      for (int e = 0; e < 4; ++e) af[12 + e] = (__bf16)lrelu(av + c3[e]);
      union { v16bf v; v8bf h[2]; } bf;
      bf.h[0] = *(const v8bf*)(&Wt[cur][m][j0 + kb]);
      bf.h[1] = *(const v8bf*)(&Wt[cur][m][j0 + kb + 16]);
      acc = __builtin_amdgcn_wmma_f32_16x16x32_bf16(false, af, false, bf.v,
                                                    (short)0, acc, false, false);
    }
    __syncthreads();
  }

  if (m < 8) {
    int rbase = (lane >> 4) * 8;
    float bv = e2n_b[m];
#pragma unroll
    for (int r = 0; r < 8; ++r) {
      float v = lrelu(acc[r] + bv);
      nodes[((size_t)b * DD + i_base + rbase + r) * 8 + m] = v;
    }
  }
}

// ---------------- stage 3: GCN attention, top-k pooling, readout, MLP ----------------
__global__ void __launch_bounds__(256) k_graph(
    const float* __restrict__ x, const float* __restrict__ nodes_g,
    const float* __restrict__ a11_w, const float* __restrict__ a11_b,
    const float* __restrict__ a12_w, const float* __restrict__ a12_b,
    const float* __restrict__ g1_w,  const float* __restrict__ g1_b,
    const float* __restrict__ a21_w, const float* __restrict__ a21_b,
    const float* __restrict__ a22_w, const float* __restrict__ a22_b,
    const float* __restrict__ d1_w,  const float* __restrict__ d1_b,
    const float* __restrict__ d2_w,  const float* __restrict__ d2_b,
    const float* __restrict__ d3_w,  const float* __restrict__ d3_b,
    float* __restrict__ outp)
{
  __shared__ float nodes[DD][8];
  __shared__ float rdeg[DD];
  __shared__ float tt[DD];
  __shared__ float sc[DD];
  __shared__ int   idx1[NK1];
  __shared__ float s1v[NK1];
  __shared__ float out1[NK1][8];
  __shared__ unsigned g1m[NK1][NK1 / 32];
  __shared__ float rdeg1[NK1];
  __shared__ float h2s[NK1][8];
  __shared__ float tt2[NK1];
  __shared__ float sc2[NK1];
  __shared__ int   idx2[NK2];
  __shared__ float s2v[NK2];
  __shared__ float zred[16];
  __shared__ float z1[128];
  __shared__ float z2[64];

  int b = blockIdx.x, t = threadIdx.x;
  const float* adj = x + (size_t)b * DD * DD;

  for (int f = 0; f < 8; ++f) nodes[t][f] = nodes_g[((size_t)b * DD + t) * 8 + f];
  {
    int d = 0;
    const float* row = adj + (size_t)t * DD;
    for (int j = 0; j < DD; ++j) d += (row[j] > 0.f);
    rdeg[t] = (d > 0) ? 1.f / (float)d : 0.f;
  }
  __syncthreads();

  { float s = 0.f; for (int f = 0; f < 8; ++f) s += nodes[t][f] * a11_w[f]; tt[t] = s * rdeg[t]; }
  __syncthreads();
  {
    float s = 0.f;
    const float* row = adj + (size_t)t * DD;
    for (int j = 0; j < DD; ++j) if (row[j] > 0.f) s += tt[j];
    sc[t] = lrelu(s + a11_b[0]);
  }
  __syncthreads();
  tt[t] = sc[t] * a12_w[0] * rdeg[t];
  __syncthreads();
  {
    float s = 0.f;
    const float* row = adj + (size_t)t * DD;
    for (int j = 0; j < DD; ++j) if (row[j] > 0.f) s += tt[j];
    sc[t] = sigm(s + a12_b[0]);
  }
  __syncthreads();

  {
    float v = sc[t]; int r = 0;
    for (int j = 0; j < DD; ++j) { float u = sc[j]; r += (u > v) || (u == v && j < t); }
    if (r < NK1) { idx1[r] = t; s1v[r] = v; }
  }
  __syncthreads();
  if (t < NK1) {
    int srcn = idx1[t]; float sv = 1.f + s1v[t];
    for (int f = 0; f < 8; ++f) out1[t][f] = nodes[srcn][f] * sv;
  }
  __syncthreads();

  if (t < 16) {
    int f = t & 7;
    if (t < 8) { float mx = -INFINITY; for (int r = 0; r < NK1; ++r) mx = fmaxf(mx, out1[r][f]); zred[t] = mx; }
    else       { float s = 0.f;        for (int r = 0; r < NK1; ++r) s += out1[r][f];          zred[t] = s / (float)NK1; }
  }
  if (t < NK1) {
    const float* row = adj + (size_t)idx1[t] * DD;
    int d = 0; unsigned w[4] = {0u, 0u, 0u, 0u};
    for (int c = 0; c < NK1; ++c) {
      bool e = row[idx1[c]] > 0.f;
      d += e;
      if (e) w[c >> 5] |= 1u << (c & 31);
    }
    for (int q = 0; q < 4; ++q) g1m[t][q] = w[q];
    rdeg1[t] = (d > 0) ? 1.f / (float)d : 0.f;
  }
  __syncthreads();

  if (t < NK1) {
    float mm[8];
    for (int f = 0; f < 8; ++f) mm[f] = 0.f;
    for (int c = 0; c < NK1; ++c)
      if ((g1m[t][c >> 5] >> (c & 31)) & 1u) {
        float rd = rdeg1[c];
        for (int f = 0; f < 8; ++f) mm[f] += out1[c][f] * rd;
      }
    for (int p = 0; p < 8; ++p) {
      float s = g1_b[p];
      for (int f = 0; f < 8; ++f) s += mm[f] * g1_w[p * 8 + f];
      h2s[t][p] = s;
    }
  }
  __syncthreads();
  if (t < NK1) { float s = 0.f; for (int f = 0; f < 8; ++f) s += h2s[t][f] * a21_w[f]; tt2[t] = s * rdeg1[t]; }
  __syncthreads();
  if (t < NK1) {
    float s = 0.f;
    for (int c = 0; c < NK1; ++c) if ((g1m[t][c >> 5] >> (c & 31)) & 1u) s += tt2[c];
    sc2[t] = lrelu(s + a21_b[0]);
  }
  __syncthreads();
  if (t < NK1) tt2[t] = sc2[t] * a22_w[0] * rdeg1[t];
  __syncthreads();
  if (t < NK1) {
    float s = 0.f;
    for (int c = 0; c < NK1; ++c) if ((g1m[t][c >> 5] >> (c & 31)) & 1u) s += tt2[c];
    sc2[t] = sigm(s + a22_b[0]);
  }
  __syncthreads();
  if (t < NK1) {
    float v = sc2[t]; int r = 0;
    for (int c = 0; c < NK1; ++c) { float u = sc2[c]; r += (u > v) || (u == v && c < t); }
    if (r < NK2) { idx2[r] = t; s2v[r] = v; }
  }
  __syncthreads();
  if (t < 16) {
    int f = t & 7;
    if (t < 8) {
      float mx = -INFINITY;
      for (int q = 0; q < NK2; ++q) mx = fmaxf(mx, h2s[idx2[q]][f] * (1.f + s2v[q]));
      zred[t] += mx;
    } else {
      float s = 0.f;
      for (int q = 0; q < NK2; ++q) s += h2s[idx2[q]][f] * (1.f + s2v[q]);
      zred[t] += s / (float)NK2;
    }
  }
  __syncthreads();
  if (t < 128) {
    float s = d1_b[t];
    for (int k = 0; k < 16; ++k) s += zred[k] * d1_w[t * 16 + k];
    z1[t] = lrelu(s);
  }
  __syncthreads();
  if (t < 64) {
    float s = d2_b[t];
    for (int k = 0; k < 128; ++k) s += z1[k] * d2_w[t * 128 + k];
    z2[t] = lrelu(s);
  }
  __syncthreads();
  if (t == 0) {
    float s = d3_b[0];
    for (int k = 0; k < 64; ++k) s += z2[k] * d3_w[k];
    outp[b] = s;
  }
}

// ---------------- launch ----------------
extern "C" void kernel_launch(void* const* d_in, const int* in_sizes, int n_in,
                              void* d_out, int out_size, void* d_ws, size_t ws_size,
                              hipStream_t stream) {
  const float* x     = (const float*)d_in[0];
  const float* w1a   = (const float*)d_in[1];
  const float* b1a   = (const float*)d_in[2];
  const float* w1b   = (const float*)d_in[3];
  const float* b1b   = (const float*)d_in[4];
  const float* w2a   = (const float*)d_in[5];
  const float* b2a   = (const float*)d_in[6];
  const float* w2b   = (const float*)d_in[7];
  const float* b2b   = (const float*)d_in[8];
  const float* e2n_w = (const float*)d_in[9];
  const float* e2n_b = (const float*)d_in[10];

  char* ws = (char*)d_ws;
  size_t off = 0;
  float*  a1      = (float*)(ws + off);  off += (size_t)BB * C1 * DD * 4;
  float*  b1      = (float*)(ws + off);  off += (size_t)BB * C1 * DD * 4;
  float*  a2      = (float*)(ws + off);  off += (size_t)BB * C2 * DD * 4;
  float*  b2      = (float*)(ws + off);  off += (size_t)BB * C2 * DD * 4;
  float*  nodes   = (float*)(ws + off);  off += (size_t)BB * DD * 8 * 4;
  __bf16* w2a_bf  = (__bf16*)(ws + off); off += (size_t)C2 * C1 * DD * 2;
  __bf16* w2b_bf  = (__bf16*)(ws + off); off += (size_t)C2 * C1 * DD * 2;
  __bf16* e2n_bf  = (__bf16*)(ws + off); off += (size_t)16 * C2 * DD * 2;

  k_e2e1<<<(2 * BB * C1 * DD) / 256, 256, 0, stream>>>(x, w1a, b1a, w1b, b1b, a1, b1);
  {
    int n = C2 * C1 * DD;
    k_cvt_bf16<<<(n + 255) / 256, 256, 0, stream>>>(w2a, w2a_bf, n);
    k_cvt_bf16<<<(n + 255) / 256, 256, 0, stream>>>(w2b, w2b_bf, n);
    k_cvt_e2n<<<(16 * C2 * DD) / 256, 256, 0, stream>>>(e2n_w, e2n_bf);
  }

  // dominant GEMMs: dynamic LDS = 2 W buffers (64 x 264 bf16) + 2 col buffers
  size_t smem = (size_t)2 * 64 * WPAD * sizeof(__bf16) + 2 * 256 * sizeof(float);
  k_e2e2_wmma<<<dim3(8, BB, 2), 256, smem, stream>>>(a1, b1, w2a_bf, w2b_bf,
                                                     b2a, b2b, a2, b2);

  k_e2n_wmma<<<dim3(2, BB), 256, 0, stream>>>(a2, b2, e2n_bf, e2n_b, nodes);

  k_graph<<<BB, 256, 0, stream>>>(
      x, nodes,
      (const float*)d_in[11], (const float*)d_in[12],
      (const float*)d_in[13], (const float*)d_in[14],
      (const float*)d_in[15], (const float*)d_in[16],
      (const float*)d_in[17], (const float*)d_in[18],
      (const float*)d_in[19], (const float*)d_in[20],
      (const float*)d_in[21], (const float*)d_in[22],
      (const float*)d_in[23], (const float*)d_in[24],
      (const float*)d_in[25], (const float*)d_in[26],
      (float*)d_out);
}
